// ISTFT_58016418234390
// MI455X (gfx1250) — compile-verified, matrix operand order checked by reading, not run on hardware
//
#include <hip/hip_runtime.h>

typedef __attribute__((ext_vector_type(16))) __bf16 v16bf;
typedef __attribute__((ext_vector_type(8)))  float  v8f;

#define B_SZ   16
#define CUTOFF 1025
#define TT     1024          // frames (T)
#define NFFT   2048
#define HOP    512
#define KDIM   2050          // 2*cutoff
#define KPAD   2080          // 65 * 32, zero-padded K
#define KSTEPS (KPAD / 32)   // 65
#define NSIG   (NFFT + HOP * (TT - 1))   // 525824
#define HALFW  (NFFT / 2)                // 1024
#define OUTLEN (NSIG - NFFT)             // 523776 per batch

#define TM  128
#define TN  128
#define KT  32
#define LDA 40               // padded LDS row stride (bf16 elems): 80B, 16B-aligned, conflict-free
#define LDS_BUF_BYTES (TM * LDA * 2)     // 10240 bytes per plane per buffer

// ---- scalar bf16 helpers (bit-exact RNE) ----
__device__ __forceinline__ unsigned short f2bf(float f) {
  unsigned u = __float_as_uint(f);
  unsigned r = u + 0x7fffu + ((u >> 16) & 1u);
  return (unsigned short)(r >> 16);
}
__device__ __forceinline__ float bf2f(unsigned short h) {
  return __uint_as_float(((unsigned)h) << 16);
}

// Async 16B global -> LDS copy (CDNA5 TDM-adjacent path, ASYNCcnt-tracked).
// ldsoff = 32-bit LDS byte address (low 32 bits of generic LDS pointer),
// g      = 64-bit global address.
#define ASYNC_CP16(ldsoff, g)                                            \
  asm volatile("global_load_async_to_lds_b128 %0, %1, off"               \
               :: "v"(ldsoff), "v"(g) : "memory")
#define WAIT_ASYNC0() asm volatile("s_wait_asynccnt 0x0" ::: "memory")

// 16-bf16 WMMA fragment from an LDS row (CDNA5 ISA 16-bit layout):
// lane<16: elems 0..7 = K 0..7,   8..15 = K 16..23
// lane>=16: elems 0..7 = K 8..15, 8..15 = K 24..31
__device__ __forceinline__ v16bf load_frag(const unsigned short* row, int half) {
  union { uint4 q[2]; v16bf v; } u;
  u.q[0] = *(const uint4*)(row + half * 8);
  u.q[1] = *(const uint4*)(row + 16 + half * 8);
  return u.v;
}

__global__ __launch_bounds__(256) void zero_kernel(float* __restrict__ p, int n) {
  int i = blockIdx.x * 256 + threadIdx.x;
  if (i < n) p[i] = 0.f;
}

// inverse_basis [2050, 2048] (c-major) -> k-major bf16 hi/lo planes [2048][KPAD]
__global__ __launch_bounds__(256) void prep_basis_kernel(const float* __restrict__ inv,
                                                         unsigned short* __restrict__ wth,
                                                         unsigned short* __restrict__ wtl) {
  int idx = blockIdx.x * 256 + threadIdx.x;
  if (idx >= KPAD * NFFT) return;
  int c = idx >> 11;            // / 2048 -> 0..2079
  int k = idx & (NFFT - 1);     // contiguous reads over k
  float v = (c < KDIM) ? inv[(size_t)c * NFFT + k] : 0.f;
  unsigned short h = f2bf(v);
  size_t o = (size_t)k * KPAD + c;
  wth[o] = h;
  wtl[o] = f2bf(v - bf2f(h));
}

// x [B,1025,1024,2] -> transposed bf16 hi/lo planes [b][t][KPAD] (one-time split)
__global__ __launch_bounds__(256) void prep_x_kernel(const float* __restrict__ x,
                                                     unsigned short* __restrict__ xth,
                                                     unsigned short* __restrict__ xtl) {
  int idx = blockIdx.x * 256 + threadIdx.x;
  if (idx >= B_SZ * KPAD * TT) return;
  int b = idx / (KPAD * TT);
  int r = idx - b * (KPAD * TT);
  int c = r / TT;               // 0..2079
  int t = r - c * TT;           // contiguous reads over t (stride-2 f32)
  float v = 0.f;
  if (c < KDIM) {
    int chn = (c >= CUTOFF) ? 1 : 0;
    int cc = c - chn * CUTOFF;
    v = x[(((size_t)b * CUTOFF + cc) * TT + t) * 2 + chn];
  }
  unsigned short h = f2bf(v);
  size_t o = ((size_t)b * TT + t) * KPAD + c;
  xth[o] = h;
  xtl[o] = f2bf(v - bf2f(h));
}

// frames GEMM (split-bf16 x3, f32 accumulate), async-copy double-buffered LDS,
// fused overlap-add epilogue
__global__ __launch_bounds__(256) void istft_gemm_kernel(const unsigned short* __restrict__ xth,
                                                         const unsigned short* __restrict__ xtl,
                                                         const unsigned short* __restrict__ wth,
                                                         const unsigned short* __restrict__ wtl,
                                                         float* __restrict__ acc) {
  __shared__ __align__(16) unsigned short sAh[2][TM * LDA];
  __shared__ __align__(16) unsigned short sAl[2][TM * LDA];
  __shared__ __align__(16) unsigned short sBh[2][TN * LDA];
  __shared__ __align__(16) unsigned short sBl[2][TN * LDA];

  const int b    = blockIdx.z;
  const int t0   = blockIdx.y * TM;
  const int k0   = blockIdx.x * TN;
  const int tid  = threadIdx.x;
  const int wave = tid >> 5;          // wave32
  const int lane = tid & 31;
  const int wr   = wave >> 1;         // 0..3 -> t offset wr*32
  const int wc   = wave & 1;          // 0..1 -> k offset wc*64
  const int half = lane >> 4;
  const int mn   = lane & 15;

  // per-thread copy slots: rows row0,row1 x {Ah,Al,Bh,Bl}, 16B each
  const int row0 = tid >> 2;          // 0..63
  const int row1 = row0 + 64;         // 64..127
  const int q16  = (tid & 3) * 8;     // short offset within 64B row

  const unsigned short* gA0h = xth + ((size_t)(b * TT + t0) + row0) * KPAD + q16;
  const unsigned short* gA1h = xth + ((size_t)(b * TT + t0) + row1) * KPAD + q16;
  const unsigned short* gA0l = xtl + ((size_t)(b * TT + t0) + row0) * KPAD + q16;
  const unsigned short* gA1l = xtl + ((size_t)(b * TT + t0) + row1) * KPAD + q16;
  const unsigned short* gB0h = wth + ((size_t)(k0 + row0)) * KPAD + q16;
  const unsigned short* gB1h = wth + ((size_t)(k0 + row1)) * KPAD + q16;
  const unsigned short* gB0l = wtl + ((size_t)(k0 + row0)) * KPAD + q16;
  const unsigned short* gB1l = wtl + ((size_t)(k0 + row1)) * KPAD + q16;

  // LDS byte addresses (buffer 0); buffer 1 is +LDS_BUF_BYTES within each plane
  const unsigned lA0h = (unsigned)(size_t)&sAh[0][row0 * LDA + q16];
  const unsigned lA1h = (unsigned)(size_t)&sAh[0][row1 * LDA + q16];
  const unsigned lA0l = (unsigned)(size_t)&sAl[0][row0 * LDA + q16];
  const unsigned lA1l = (unsigned)(size_t)&sAl[0][row1 * LDA + q16];
  const unsigned lB0h = (unsigned)(size_t)&sBh[0][row0 * LDA + q16];
  const unsigned lB1h = (unsigned)(size_t)&sBh[0][row1 * LDA + q16];
  const unsigned lB0l = (unsigned)(size_t)&sBl[0][row0 * LDA + q16];
  const unsigned lB1l = (unsigned)(size_t)&sBl[0][row1 * LDA + q16];

  auto issue = [&](int ksn, int nbuf) {
    const size_t cb = (size_t)ksn * KT;           // element (short) offset in K
    const unsigned lo = (unsigned)(nbuf * LDS_BUF_BYTES);
    ASYNC_CP16(lA0h + lo, gA0h + cb);
    ASYNC_CP16(lA1h + lo, gA1h + cb);
    ASYNC_CP16(lA0l + lo, gA0l + cb);
    ASYNC_CP16(lA1l + lo, gA1l + cb);
    ASYNC_CP16(lB0h + lo, gB0h + cb);
    ASYNC_CP16(lB1h + lo, gB1h + cb);
    ASYNC_CP16(lB0l + lo, gB0l + cb);
    ASYNC_CP16(lB1l + lo, gB1l + cb);
  };

  v8f vzero = {};
  v8f accv[2][4];
#pragma unroll
  for (int i = 0; i < 2; ++i)
#pragma unroll
    for (int j = 0; j < 4; ++j) accv[i][j] = vzero;

  issue(0, 0);   // prologue: batch 0 -> buffer 0

  for (int ks = 0; ks < KSTEPS; ++ks) {
    const int buf = ks & 1;
    WAIT_ASYNC0();        // my chunks of batch ks are in LDS
    __syncthreads();      // everyone's chunks in; everyone done reading buf^1
    if (ks + 1 < KSTEPS) issue(ks + 1, buf ^ 1);   // overlaps with WMMAs below

    // fragments + WMMA (split-bf16: hi*hi + hi*lo + lo*hi)
    v16bf ah[2], al[2];
#pragma unroll
    for (int i = 0; i < 2; ++i) {
      int arow = wr * 32 + i * 16 + mn;
      ah[i] = load_frag(&sAh[buf][arow * LDA], half);
      al[i] = load_frag(&sAl[buf][arow * LDA], half);
    }
#pragma unroll
    for (int j = 0; j < 4; ++j) {
      int nrow = wc * 64 + j * 16 + mn;
      v16bf bh = load_frag(&sBh[buf][nrow * LDA], half);
      v16bf bl = load_frag(&sBl[buf][nrow * LDA], half);
#pragma unroll
      for (int i = 0; i < 2; ++i) {
        accv[i][j] = __builtin_amdgcn_wmma_f32_16x16x32_bf16(false, ah[i], false, bh,
                                                             (short)0, accv[i][j], false, false);
        accv[i][j] = __builtin_amdgcn_wmma_f32_16x16x32_bf16(false, ah[i], false, bl,
                                                             (short)0, accv[i][j], false, false);
        accv[i][j] = __builtin_amdgcn_wmma_f32_16x16x32_bf16(false, al[i], false, bh,
                                                             (short)0, accv[i][j], false, false);
      }
    }
  }

  // Epilogue: overlap-add frames[t,k] into acc[b, t*HOP + k]
  // C/D layout: VGPR r -> M = r + 8*(lane>=16), N = lane%16
  float* accb = acc + (size_t)b * NSIG;
#pragma unroll
  for (int i = 0; i < 2; ++i) {
#pragma unroll
    for (int j = 0; j < 4; ++j) {
      int kcol = k0 + wc * 64 + j * 16 + mn;
#pragma unroll
      for (int r = 0; r < 8; ++r) {
        int trow = t0 + wr * 32 + i * 16 + r + half * 8;
        atomicAdd(accb + (size_t)trow * HOP + kcol, accv[i][j][r]);
      }
    }
  }
}

__global__ __launch_bounds__(256) void normalize_kernel(const float* __restrict__ acc,
                                                        const float* __restrict__ wsum,
                                                        float* __restrict__ out) {
  int idx = blockIdx.x * 256 + threadIdx.x;
  if (idx >= B_SZ * OUTLEN) return;
  int bb = idx / OUTLEN;
  int i  = idx - bb * OUTLEN;
  float w = wsum[i + HALFW];
  out[idx] = acc[(size_t)bb * NSIG + i + HALFW] / (w + 1e-8f) * 4.0f;
}

extern "C" void kernel_launch(void* const* d_in, const int* in_sizes, int n_in,
                              void* d_out, int out_size, void* d_ws, size_t ws_size,
                              hipStream_t stream) {
  const float* x    = (const float*)d_in[0];   // [16,1025,1024,2] f32
  const float* inv  = (const float*)d_in[1];   // [2050,2048] f32
  const float* wsum = (const float*)d_in[2];   // window_sum f32
  float* out = (float*)d_out;

  char* ws = (char*)d_ws;
  float* acc = (float*)ws;                                   // 33.7 MB
  size_t off = (size_t)B_SZ * NSIG * sizeof(float);
  unsigned short* wth = (unsigned short*)(ws + off);         // 8.5 MB
  off += (size_t)NFFT * KPAD * 2;
  unsigned short* wtl = (unsigned short*)(ws + off);         // 8.5 MB
  off += (size_t)NFFT * KPAD * 2;
  unsigned short* xth = (unsigned short*)(ws + off);         // 68.2 MB
  off += (size_t)B_SZ * TT * KPAD * 2;
  unsigned short* xtl = (unsigned short*)(ws + off);         // 68.2 MB

  int nacc = B_SZ * NSIG;
  zero_kernel<<<(nacc + 255) / 256, 256, 0, stream>>>(acc, nacc);

  int nprep = KPAD * NFFT;
  prep_basis_kernel<<<(nprep + 255) / 256, 256, 0, stream>>>(inv, wth, wtl);

  int nx = B_SZ * KPAD * TT;
  prep_x_kernel<<<(nx + 255) / 256, 256, 0, stream>>>(x, xth, xtl);

  dim3 grid(NFFT / TN, TT / TM, B_SZ);                        // 16 x 8 x 16
  istft_gemm_kernel<<<grid, 256, 0, stream>>>(xth, xtl, wth, wtl, acc);

  int nout = B_SZ * OUTLEN;
  normalize_kernel<<<(nout + 255) / 256, 256, 0, stream>>>(acc, wsum, out);

  (void)in_sizes; (void)n_in; (void)out_size; (void)ws_size;
}
